// EGNNMessageLayer_18133351924499
// MI455X (gfx1250) — matrix operand order repaired and checked by reading, not compiled, not run on hardware
//
#include <hip/hip_runtime.h>
#include <hip/hip_bf16.h>

// ---------------------------------------------------------------------------
// EGNN message layer for MI455X (gfx1250): bf16 WMMA (v_wmma_f32_16x16x32_bf16)
// for the edge/node MLPs, f32 atomics for scatter/graph-norm reductions.
// Round 4: silu via v_rcp_f32, graph-sum accumulation fused into the node
// kernel epilogue (drops one full d_out read pass).
// ---------------------------------------------------------------------------

#define NN   20000
#define EE   320000
#define GG   64
#define DSZ  256   // source feat
#define DTT  256   // target feat
#define DDD  64    // distance feat
#define DHH  256   // hidden
#define DOO  256   // out

typedef __attribute__((ext_vector_type(16))) __bf16 v16bf;
typedef __attribute__((ext_vector_type(8)))  __bf16 v8bf;
typedef __attribute__((ext_vector_type(4)))  __bf16 v4bf;
typedef __attribute__((ext_vector_type(8)))  float  v8f;
typedef __attribute__((ext_vector_type(4)))  float  vf4;

union V16U { v16bf v; v8bf h[2]; };

// ---- A fragment (16x32 bf16, M x K) per ISA 7.12.2 -------------------------
// lanes 0-15 : row M=lane,    elems 0..7 -> K 0..7,  elems 8..15 -> K 16..23
// lanes 16-31: row M=lane-16, elems 0..7 -> K 8..15, elems 8..15 -> K 24..31
__device__ __forceinline__ v16bf load_a_frag(const __bf16* base, int rowStride, int lane) {
    const int r  = lane & 15;
    const int hk = (lane >> 4) << 3;      // 0 or 8
    const __bf16* p = base + (size_t)r * rowStride + hk;
    V16U u;
    u.h[0] = *reinterpret_cast<const v8bf*>(p);
    u.h[1] = *reinterpret_cast<const v8bf*>(p + 16);
    return u.v;
}

// ---- B fragment (32x16 bf16, K x N): pre-packed so each lane's 16 elems are
// contiguous (32B). lanes 0-15: K=e (e<8), 16+e-8 (e>=8), N=lane; lanes 16-31: K=8+e (e<8), 24+e-8 (e>=8), N=lane-16.
__device__ __forceinline__ v16bf load_b_frag(const __bf16* Wp, int kb, int ct, int lane) {
    return *reinterpret_cast<const v16bf*>(Wp + ((((size_t)kb * 16 + ct) * 32 + lane) << 4));
}

// silu via hardware reciprocal (single v_rcp_f32 instead of precise-div chain)
__device__ __forceinline__ float silu(float x) {
    return x * __builtin_amdgcn_rcpf(1.0f + __expf(-x));
}

// ---------------------------------------------------------------------------
// K0: pack f32 weight [K x 256] row-major -> bf16 B-fragment layout
// ---------------------------------------------------------------------------
__global__ __launch_bounds__(256) void pack_weight(const float* __restrict__ W,
                                                   __bf16* __restrict__ P,
                                                   int K, int Nc) {
    size_t idx = (size_t)blockIdx.x * 256 + threadIdx.x;
    size_t total = (size_t)K * Nc;
    if (idx >= total) return;
    int e    = (int)(idx & 15);
    int lane = (int)((idx >> 4) & 31);
    size_t t = idx >> 9;
    int nt   = Nc >> 4;
    int ct   = (int)(t % nt);
    int kb   = (int)(t / nt);
    int k = kb * 32 + ((lane >> 4) << 3) + (e < 8 ? e : e + 8);
    int n = ct * 16 + (lane & 15);
    P[idx] = (__bf16)W[(size_t)k * Nc + n];
}

// ---------------------------------------------------------------------------
// K1: edge message MLP + scatter-add.
// Block = 256 thr = 8 waves, 64 edges. Wave w owns all 4 row-tiles and
// col-tiles {2w, 2w+1}: each of the 16 B fragments per K-step is loaded by
// exactly one wave. Staging: thread owns rows r0=tid>>3 and r0+32, k-quad
// (tid&7)*4 -> two float4 gathers per K-step, pipelined one step ahead.
// ---------------------------------------------------------------------------
__global__ __launch_bounds__(256) void edge_msg_kernel(
    const float* __restrict__ src, const float* __restrict__ tgt,
    const int* __restrict__ eidx, const float* __restrict__ dist,
    const __bf16* __restrict__ W1p, const float* __restrict__ b1,
    const __bf16* __restrict__ W2p, const float* __restrict__ b2,
    float* __restrict__ aggr)
{
    __shared__ __bf16 Atile[64 * 32];      // 4 KB staging tile (one K-step)
    __shared__ __bf16 Mtile[64 * 256];     // 32 KB layer-1 activations
    __shared__ int isrc_s[64], itgt_s[64];

    const int tid  = threadIdx.x;
    const int lane = tid & 31;
    const int w    = tid >> 5;             // wave id: col-tiles 2w, 2w+1
    const int e0   = blockIdx.x * 64;

    if (tid < 64) {
        isrc_s[tid] = eidx[e0 + tid];
        itgt_s[tid] = eidx[EE + e0 + tid];
    }
    __syncthreads();

    // register-hoisted gather row pointers (loop-invariant)
    const int r0 = tid >> 3;          // 0..31
    const int r1 = r0 + 32;           // 32..63
    const int kq = (tid & 7) << 2;    // 0,4,...,28
    const float* __restrict__ srcR0 = src + (size_t)isrc_s[r0] * DSZ;
    const float* __restrict__ srcR1 = src + (size_t)isrc_s[r1] * DSZ;
    const float* __restrict__ tgtR0 = tgt + (size_t)itgt_s[r0] * DTT;
    const float* __restrict__ tgtR1 = tgt + (size_t)itgt_s[r1] * DTT;
    const float* __restrict__ dstR0 = dist + (size_t)(e0 + r0) * DDD;
    const float* __restrict__ dstR1 = dist + (size_t)(e0 + r1) * DDD;

    auto gather0 = [&](int kb) -> vf4 {
        int k = kb * 32 + kq;
        const float* p = (k < DSZ) ? (srcR0 + k)
                       : (k < DSZ + DTT) ? (tgtR0 + (k - DSZ))
                       : (dstR0 + (k - DSZ - DTT));
        return *reinterpret_cast<const vf4*>(p);
    };
    auto gather1 = [&](int kb) -> vf4 {
        int k = kb * 32 + kq;
        const float* p = (k < DSZ) ? (srcR1 + k)
                       : (k < DSZ + DTT) ? (tgtR1 + (k - DSZ))
                       : (dstR1 + (k - DSZ - DTT));
        return *reinterpret_cast<const vf4*>(p);
    };

    v8f acc[8];   // [cc*4 + rt]
#pragma unroll
    for (int i = 0; i < 8; ++i) acc[i] = (v8f){0.f,0.f,0.f,0.f,0.f,0.f,0.f,0.f};

    // ---------------- layer 1: K = DS+DT+DD = 576 (18 steps of 32) ----------
    vf4 c0 = gather0(0);
    vf4 c1 = gather1(0);
    for (int kb = 0; kb < 18; ++kb) {
        __syncthreads();   // previous compute done reading Atile
        *reinterpret_cast<v4bf*>(&Atile[r0 * 32 + kq]) = __builtin_convertvector(c0, v4bf);
        *reinterpret_cast<v4bf*>(&Atile[r1 * 32 + kq]) = __builtin_convertvector(c1, v4bf);
        int kn = (kb + 1 < 18) ? kb + 1 : 17;    // prefetch next step early
        vf4 n0 = gather0(kn);
        vf4 n1 = gather1(kn);
        __syncthreads();   // Atile ready
        v16bf a[4];
#pragma unroll
        for (int rt = 0; rt < 4; ++rt)
            a[rt] = load_a_frag(Atile + rt * 16 * 32, 32, lane);
#pragma unroll
        for (int cc = 0; cc < 2; ++cc) {
            v16bf b = load_b_frag(W1p, kb, w * 2 + cc, lane);
#pragma unroll
            for (int rt = 0; rt < 4; ++rt)
                acc[cc * 4 + rt] = __builtin_amdgcn_wmma_f32_16x16x32_bf16(
                    false, a[rt], false, b, (short)0, acc[cc * 4 + rt], false, false);
        }
        c0 = n0; c1 = n1;
    }

    // bias + silu -> Mtile (bf16), reset accumulators
    {
        const int rbg = (lane >> 4) << 3;   // 0 or 8
#pragma unroll
        for (int cc = 0; cc < 2; ++cc) {
            int col = (w * 2 + cc) * 16 + (lane & 15);
            float bb = b1[col];
#pragma unroll
            for (int rt = 0; rt < 4; ++rt) {
#pragma unroll
                for (int g = 0; g < 8; ++g) {
                    float x = acc[cc * 4 + rt][g] + bb;
                    Mtile[(size_t)(rt * 16 + rbg + g) * 256 + col] = (__bf16)silu(x);
                }
                acc[cc * 4 + rt] = (v8f){0.f,0.f,0.f,0.f,0.f,0.f,0.f,0.f};
            }
        }
    }
    __syncthreads();

    // ---------------- layer 2: K = DH = 256 (8 steps, LDS-resident) ---------
    for (int kb = 0; kb < 8; ++kb) {
        v16bf a[4];
#pragma unroll
        for (int rt = 0; rt < 4; ++rt)
            a[rt] = load_a_frag(Mtile + (size_t)rt * 16 * 256 + kb * 32, 256, lane);
#pragma unroll
        for (int cc = 0; cc < 2; ++cc) {
            v16bf b = load_b_frag(W2p, kb, w * 2 + cc, lane);
#pragma unroll
            for (int rt = 0; rt < 4; ++rt)
                acc[cc * 4 + rt] = __builtin_amdgcn_wmma_f32_16x16x32_bf16(
                    false, a[rt], false, b, (short)0, acc[cc * 4 + rt], false, false);
        }
    }

    // bias + silu + scatter-add into aggr[tgt]
    {
        const int rbg = (lane >> 4) << 3;
#pragma unroll
        for (int cc = 0; cc < 2; ++cc) {
            int col = (w * 2 + cc) * 16 + (lane & 15);
            float bb = b2[col];
#pragma unroll
            for (int rt = 0; rt < 4; ++rt) {
#pragma unroll
                for (int g = 0; g < 8; ++g) {
                    float x = silu(acc[cc * 4 + rt][g] + bb);
                    int r = rt * 16 + rbg + g;
                    atomicAdd(&aggr[(size_t)itgt_s[r] * DHH + col], x);
                }
            }
        }
    }
}

// ---------------------------------------------------------------------------
// K2: node combine MLP + residual -> pre-norm output (d_out) + fused per-graph
// sum accumulation for GraphNorm.
// ---------------------------------------------------------------------------
__global__ __launch_bounds__(256) void node_combine_kernel(
    const float* __restrict__ tgt, const float* __restrict__ aggr,
    const __bf16* __restrict__ Wc1p, const float* __restrict__ bc1,
    const __bf16* __restrict__ Wc2p, const float* __restrict__ bc2,
    const int* __restrict__ batch, float* __restrict__ gsum,
    float* __restrict__ outp)
{
    __shared__ __bf16 Atile[64 * 32];
    __shared__ __bf16 Mtile[64 * 256];

    const int tid  = threadIdx.x;
    const int lane = tid & 31;
    const int w    = tid >> 5;
    const int n0   = blockIdx.x * 64;

    // register-hoisted gather row pointers (tail rows clamped for reads)
    const int r0 = tid >> 3;
    const int r1 = r0 + 32;
    const int kq = (tid & 7) << 2;
    const int nA = (n0 + r0 < NN) ? n0 + r0 : NN - 1;
    const int nB = (n0 + r1 < NN) ? n0 + r1 : NN - 1;
    const float* __restrict__ tgtR0 = tgt  + (size_t)nA * DTT;
    const float* __restrict__ tgtR1 = tgt  + (size_t)nB * DTT;
    const float* __restrict__ agrR0 = aggr + (size_t)nA * DHH;
    const float* __restrict__ agrR1 = aggr + (size_t)nB * DHH;

    auto gather0 = [&](int kb) -> vf4 {
        int k = kb * 32 + kq;
        const float* p = (k < DTT) ? (tgtR0 + k) : (agrR0 + (k - DTT));
        return *reinterpret_cast<const vf4*>(p);
    };
    auto gather1 = [&](int kb) -> vf4 {
        int k = kb * 32 + kq;
        const float* p = (k < DTT) ? (tgtR1 + k) : (agrR1 + (k - DTT));
        return *reinterpret_cast<const vf4*>(p);
    };

    v8f acc[8];
#pragma unroll
    for (int i = 0; i < 8; ++i) acc[i] = (v8f){0.f,0.f,0.f,0.f,0.f,0.f,0.f,0.f};

    // ---------------- layer c1: K = DT+DH = 512 (16 steps) ------------------
    vf4 c0 = gather0(0);
    vf4 c1 = gather1(0);
    for (int kb = 0; kb < 16; ++kb) {
        __syncthreads();
        *reinterpret_cast<v4bf*>(&Atile[r0 * 32 + kq]) = __builtin_convertvector(c0, v4bf);
        *reinterpret_cast<v4bf*>(&Atile[r1 * 32 + kq]) = __builtin_convertvector(c1, v4bf);
        int kn = (kb + 1 < 16) ? kb + 1 : 15;
        vf4 p0 = gather0(kn);
        vf4 p1 = gather1(kn);
        __syncthreads();
        v16bf a[4];
#pragma unroll
        for (int rt = 0; rt < 4; ++rt)
            a[rt] = load_a_frag(Atile + rt * 16 * 32, 32, lane);
#pragma unroll
        for (int cc = 0; cc < 2; ++cc) {
            v16bf b = load_b_frag(Wc1p, kb, w * 2 + cc, lane);
#pragma unroll
            for (int rt = 0; rt < 4; ++rt)
                acc[cc * 4 + rt] = __builtin_amdgcn_wmma_f32_16x16x32_bf16(
                    false, a[rt], false, b, (short)0, acc[cc * 4 + rt], false, false);
        }
        c0 = p0; c1 = p1;
    }

    {
        const int rbg = (lane >> 4) << 3;
#pragma unroll
        for (int cc = 0; cc < 2; ++cc) {
            int col = (w * 2 + cc) * 16 + (lane & 15);
            float bb = bc1[col];
#pragma unroll
            for (int rt = 0; rt < 4; ++rt) {
#pragma unroll
                for (int g = 0; g < 8; ++g) {
                    float x = acc[cc * 4 + rt][g] + bb;
                    Mtile[(size_t)(rt * 16 + rbg + g) * 256 + col] = (__bf16)silu(x);
                }
                acc[cc * 4 + rt] = (v8f){0.f,0.f,0.f,0.f,0.f,0.f,0.f,0.f};
            }
        }
    }
    __syncthreads();

    // ---------------- layer c2: K = DH = 256 (8 steps, no activation) -------
    for (int kb = 0; kb < 8; ++kb) {
        v16bf a[4];
#pragma unroll
        for (int rt = 0; rt < 4; ++rt)
            a[rt] = load_a_frag(Mtile + (size_t)rt * 16 * 256 + kb * 32, 256, lane);
#pragma unroll
        for (int cc = 0; cc < 2; ++cc) {
            v16bf b = load_b_frag(Wc2p, kb, w * 2 + cc, lane);
#pragma unroll
            for (int rt = 0; rt < 4; ++rt)
                acc[cc * 4 + rt] = __builtin_amdgcn_wmma_f32_16x16x32_bf16(
                    false, a[rt], false, b, (short)0, acc[cc * 4 + rt], false, false);
        }
    }

    // bias + residual -> d_out (pre-norm), fused per-graph sum accumulation
    {
        const int rbg = (lane >> 4) << 3;
#pragma unroll
        for (int cc = 0; cc < 2; ++cc) {
            int col = (w * 2 + cc) * 16 + (lane & 15);
            float bb = bc2[col];
#pragma unroll
            for (int rt = 0; rt < 4; ++rt) {
#pragma unroll
                for (int g = 0; g < 8; ++g) {
                    int node = n0 + rt * 16 + rbg + g;
                    if (node < NN) {
                        float o = tgt[(size_t)node * DOO + col] + acc[cc * 4 + rt][g] + bb;
                        outp[(size_t)node * DOO + col] = o;
                        atomicAdd(&gsum[(size_t)batch[node] * DOO + col], o);
                    }
                }
            }
        }
    }
}

// ---------------------------------------------------------------------------
// GraphNorm: count, center + var, scale (in place on d_out), float4.
// ---------------------------------------------------------------------------
__global__ __launch_bounds__(256) void gn_count(const int* __restrict__ batch,
                                                float* __restrict__ gcnt) {
    int i = blockIdx.x * 256 + threadIdx.x;
    if (i >= NN) return;
    atomicAdd(&gcnt[batch[i]], 1.0f);
}

__global__ __launch_bounds__(256) void gn_center(float* __restrict__ x,
                                                 const int* __restrict__ batch,
                                                 const float* __restrict__ gsum,
                                                 const float* __restrict__ gcnt,
                                                 const float* __restrict__ mean_scale,
                                                 float* __restrict__ gvar) {
    size_t i = (size_t)blockIdx.x * 256 + threadIdx.x;
    if (i >= (size_t)NN * DOO / 4) return;
    size_t base = i << 2;
    int node = (int)(base >> 8), c = (int)(base & 255);
    int g = batch[node];
    float inv = __builtin_amdgcn_rcpf(fmaxf(gcnt[g], 1.0f));
    vf4 v  = *reinterpret_cast<const vf4*>(x + base);
    vf4 ms = *reinterpret_cast<const vf4*>(mean_scale + c);
    vf4 sm = *reinterpret_cast<const vf4*>(gsum + (size_t)g * DOO + c);
    vf4 o  = v - sm * inv * ms;
    *reinterpret_cast<vf4*>(x + base) = o;
    float* gv = gvar + (size_t)g * DOO + c;
    atomicAdd(gv + 0, o.x * o.x);
    atomicAdd(gv + 1, o.y * o.y);
    atomicAdd(gv + 2, o.z * o.z);
    atomicAdd(gv + 3, o.w * o.w);
}

__global__ __launch_bounds__(256) void gn_scale(float* __restrict__ x,
                                                const int* __restrict__ batch,
                                                const float* __restrict__ gvar,
                                                const float* __restrict__ gcnt,
                                                const float* __restrict__ wgt,
                                                const float* __restrict__ bias) {
    size_t i = (size_t)blockIdx.x * 256 + threadIdx.x;
    if (i >= (size_t)NN * DOO / 4) return;
    size_t base = i << 2;
    int node = (int)(base >> 8), c = (int)(base & 255);
    int g = batch[node];
    float inv = __builtin_amdgcn_rcpf(fmaxf(gcnt[g], 1.0f));
    vf4 v  = *reinterpret_cast<const vf4*>(x + base);
    vf4 vr = *reinterpret_cast<const vf4*>(gvar + (size_t)g * DOO + c);
    vf4 wv = *reinterpret_cast<const vf4*>(wgt + c);
    vf4 bv = *reinterpret_cast<const vf4*>(bias + c);
    vf4 o;
    o.x = wv.x * v.x * rsqrtf(vr.x * inv + 1e-5f) + bv.x;
    o.y = wv.y * v.y * rsqrtf(vr.y * inv + 1e-5f) + bv.y;
    o.z = wv.z * v.z * rsqrtf(vr.z * inv + 1e-5f) + bv.z;
    o.w = wv.w * v.w * rsqrtf(vr.w * inv + 1e-5f) + bv.w;
    *reinterpret_cast<vf4*>(x + base) = o;
}

// ---------------------------------------------------------------------------
extern "C" void kernel_launch(void* const* d_in, const int* in_sizes, int n_in,
                              void* d_out, int out_size, void* d_ws, size_t ws_size,
                              hipStream_t stream) {
    const float* src  = (const float*)d_in[0];
    const float* tgt  = (const float*)d_in[1];
    const int*   eidx = (const int*)d_in[2];
    // d_in[3] = edge_attr (size 0, unused)
    const float* dist = (const float*)d_in[4];
    const int*   bat  = (const int*)d_in[5];
    const float* W1   = (const float*)d_in[6];
    const float* b1   = (const float*)d_in[7];
    const float* W2   = (const float*)d_in[8];
    const float* b2   = (const float*)d_in[9];
    const float* Wc1  = (const float*)d_in[10];
    const float* bc1  = (const float*)d_in[11];
    const float* Wc2  = (const float*)d_in[12];
    const float* bc2  = (const float*)d_in[13];
    const float* gw   = (const float*)d_in[14];
    const float* gb   = (const float*)d_in[15];
    const float* gms  = (const float*)d_in[16];
    float* out = (float*)d_out;

    char* ws = (char*)d_ws;
    size_t off = 0;
    auto take = [&](size_t bytes) -> void* {
        void* p = (void*)(ws + off);
        off += (bytes + 255) & ~(size_t)255;
        return p;
    };
    __bf16* W1p  = (__bf16*)take((size_t)(DSZ + DTT + DDD) * DHH * 2);
    __bf16* W2p  = (__bf16*)take((size_t)DHH * DHH * 2);
    __bf16* Wc1p = (__bf16*)take((size_t)(DTT + DHH) * DHH * 2);
    __bf16* Wc2p = (__bf16*)take((size_t)DHH * DOO * 2);
    float*  aggr = (float*)take((size_t)NN * DHH * 4);
    float*  gsum = (float*)take((size_t)GG * DOO * 4);
    float*  gvar = (float*)take((size_t)GG * DOO * 4);
    float*  gcnt = (float*)take((size_t)GG * 4);

    // zero accumulation buffers (graph-capturable async memsets)
    hipMemsetAsync(aggr, 0, (size_t)NN * DHH * 4, stream);
    hipMemsetAsync(gsum, 0, (size_t)GG * DOO * 4, stream);
    hipMemsetAsync(gvar, 0, (size_t)GG * DOO * 4, stream);
    hipMemsetAsync(gcnt, 0, (size_t)GG * 4, stream);

    // K0: weight repack to bf16 fragment layout
    pack_weight<<<((DSZ + DTT + DDD) * DHH + 255) / 256, 256, 0, stream>>>(W1, W1p, DSZ + DTT + DDD, DHH);
    pack_weight<<<(DHH * DHH + 255) / 256, 256, 0, stream>>>(W2, W2p, DHH, DHH);
    pack_weight<<<((DTT + DHH) * DHH + 255) / 256, 256, 0, stream>>>(Wc1, Wc1p, DTT + DHH, DHH);
    pack_weight<<<(DHH * DOO + 255) / 256, 256, 0, stream>>>(Wc2, Wc2p, DHH, DOO);

    // K1: edge MLP + scatter (E % 64 == 0)
    edge_msg_kernel<<<EE / 64, 256, 0, stream>>>(src, tgt, eidx, dist, W1p, b1, W2p, b2, aggr);

    // per-graph node counts (tiny, overlaps conceptually with K2's tail)
    gn_count<<<(NN + 255) / 256, 256, 0, stream>>>(bat, gcnt);

    // K2: node combine MLP + residual -> d_out, fused per-graph sums
    node_combine_kernel<<<(NN + 63) / 64, 256, 0, stream>>>(tgt, aggr, Wc1p, bc1, Wc2p, bc2, bat, gsum, out);

    // GraphNorm in place on d_out (float4 per thread)
    int gnBlocks = (NN * DOO / 4 + 255) / 256;
    gn_center<<<gnBlocks, 256, 0, stream>>>(out, bat, gsum, gcnt, gms, gvar);
    gn_scale<<<gnBlocks, 256, 0, stream>>>(out, bat, gvar, gcnt, gw, gb);
}
